// ImprovedClassForgeGNN_44736379355243
// MI455X (gfx1250) — matrix operand order repaired and checked by reading, not compiled, not run on hardware
//
#include <hip/hip_runtime.h>
#include <hip/hip_bf16.h>
#include <cstdint>

// Problem constants (match reference)
#define GN 50000
#define GE 800000
#define GIN 128
#define GH 256
#define GEMB 128
#define GR 4
#define GHEADS 2

typedef __attribute__((ext_vector_type(16))) __bf16 v16bf;
typedef __attribute__((ext_vector_type(8)))  float  v8f;
typedef __attribute__((ext_vector_type(4)))  unsigned int u32x4;
typedef __attribute__((ext_vector_type(4)))  int  i32x4;
typedef __attribute__((ext_vector_type(8)))  int  i32x8;

#if __has_builtin(__builtin_amdgcn_tensor_load_to_lds) && \
    __has_builtin(__builtin_amdgcn_s_wait_tensorcnt)
#define USE_TDM 1
#else
#define USE_TDM 0
#endif

__device__ __forceinline__ unsigned short f2bf(float f) {
  union { float f; unsigned int u; } v; v.f = f;
  unsigned int lsb = (v.u >> 16) & 1u;
  v.u += 0x7FFFu + lsb;                 // round-to-nearest-even
  return (unsigned short)(v.u >> 16);
}

__device__ __forceinline__ unsigned pack2(float a, float b) {
#if __has_builtin(__builtin_amdgcn_cvt_pk_bf16_f32)
  typedef __attribute__((ext_vector_type(2))) __bf16 bf16x2;
  union { bf16x2 v; unsigned u; } r;
  r.v = __builtin_amdgcn_cvt_pk_bf16_f32(a, b);
  return r.u;
#else
  return (unsigned)f2bf(a) | ((unsigned)f2bf(b) << 16);
#endif
}

__device__ __forceinline__ void atomicMaxF(float* addr, float val) {
  if (val >= 0.0f) atomicMax((int*)addr, __float_as_int(val));
  else             atomicMin((unsigned int*)addr, __float_as_uint(val));
}

// ---------------------------------------------------------------------------
// WMMA GEMM: C[M,N](f32) = A[M,K](f32, stride lda) x B^T  where Bt is the
// PRE-TRANSPOSED bf16 weight, layout [N][K] with row stride ldb(=K span).
// Block tile 64M x 128N (N must be a multiple of 128), 8 waves; each wave
// owns 16M x 64N -> 4 v_wmma_f32_16x16x32_bf16 per K-step with the A
// fragment reused across all four. Double-buffered software pipeline:
// while computing tile i from LDS buf cur, wave0 issues an async TDM
// (tensor_load_to_lds, TENSORcnt) for B tile i+1 into buf nxt and all waves
// stage A tile i+1 (f32->bf16). One barrier per K-step.
// flags: bit0 = accumulate into C, bit1 = relu.
// ---------------------------------------------------------------------------
__global__ __launch_bounds__(256)
void k_wmma_gemm(const float* __restrict__ A, int lda,
                 const unsigned short* __restrict__ Bt, int ldb,
                 const float* __restrict__ bias,
                 float* __restrict__ C, int ldc,
                 int M, int K, int N, int flags)
{
  __shared__ unsigned short lA[2][64][40];    // [buf][m][k], 80B rows
  __shared__ unsigned short lB[2][128][40];   // [buf][n][k] transposed

  const int tid  = threadIdx.x;
  const int lane = tid & 31;
  const int wave = tid >> 5;          // 0..7
  const int wm   = wave & 3;          // M sub-tile (16 rows each)
  const int wn   = wave >> 2;         // N sub-tile (64 cols each)
  const int half = lane >> 4;
  const int l16  = lane & 15;

  const int blockM = blockIdx.y * 64;
  const int blockN = blockIdx.x * 128;
  const int mBase  = blockM + wm * 16;
  const int nBase  = blockN + wn * 64;

  // staging coords: rows x 32 cols, 8 elems (one uint4) per slot
  const int sr = tid >> 2;            // 0..63
  const int sc = (tid & 3) * 8;       // 0,8,16,24

  // ---- stage helpers ----
  auto stageA = [&](int k0, int buf) {
    int gm = blockM + sr;
    float4 v0 = make_float4(0.f, 0.f, 0.f, 0.f);
    float4 v1 = v0;
    if (gm < M) {
      const float* ap = A + (size_t)gm * lda + k0 + sc;
      v0 = *(const float4*)ap;
      v1 = *(const float4*)(ap + 4);
      if (k0 + 32 < K) __builtin_prefetch(ap + 32, 0, 3);
    }
    uint4 p;
    p.x = pack2(v0.x, v0.y); p.y = pack2(v0.z, v0.w);
    p.z = pack2(v1.x, v1.y); p.w = pack2(v1.z, v1.w);
    *(uint4*)&lA[buf][sr][sc] = p;
  };

  auto issueB = [&](int k0, int buf) {
#if USE_TDM
    if (wave == 0) {
      const unsigned short* gB = Bt + (size_t)blockN * ldb + k0;
      unsigned long long ga = (unsigned long long)(uintptr_t)gB;
      unsigned ldsB = (unsigned)(uintptr_t)&lB[buf][0][0];
      u32x4 g0;
      g0[0] = 1u;                                    // count=1, user mode
      g0[1] = ldsB;                                  // lds_addr
      g0[2] = (unsigned)ga;                          // global_addr lo
      g0[3] = (unsigned)(ga >> 32) | (2u << 30);     // addr hi | type=2
      i32x8 g1;
      // data_size=2B (code1), pad_enable, pad_interval=16DW, pad_amount=4DW
      g1[0] = (1 << 16) | (1 << 20) | (3 << 22) | (3 << 25);
      g1[1] = (32 << 16);                            // tensor_dim0 = 32
      g1[2] = (128 << 16);                           // tensor_dim1 = 128
      g1[3] = (32 << 16);                            // tile_dim0 = 32
      g1[4] = 128;                                   // tile_dim1 = 128
      g1[5] = ldb;                                   // tensor_dim0_stride
      g1[6] = 0;
      g1[7] = 0;
      i32x4 z4 = {};
#if defined(__clang_major__) && (__clang_major__ >= 23)
      i32x8 z8 = {};
      __builtin_amdgcn_tensor_load_to_lds(g0, g1, z4, z4, z8, 0);
#else
      __builtin_amdgcn_tensor_load_to_lds(g0, g1, z4, z4, 0);
#endif
    }
#else
#pragma unroll
    for (int i = 0; i < 2; ++i) {
      int r = sr + i * 64;            // 0..127
      uint4 v = *(const uint4*)(Bt + (size_t)(blockN + r) * ldb + k0 + sc);
      *(uint4*)&lB[buf][r][sc] = v;
    }
#endif
  };

  v8f c[4] = {{}, {}, {}, {}};
  if (flags & 1) {                    // accumulate: preload existing C
#pragma unroll
    for (int j = 0; j < 4; ++j)
#pragma unroll
      for (int r = 0; r < 8; ++r) {
        int m = mBase + r + 8 * half;
        if (m < M)
          c[j][r] = C[(size_t)m * ldc + nBase + j * 16 + l16];
      }
  }

  // ---- prologue: fill buffer 0 ----
  issueB(0, 0);
  stageA(0, 0);
#if USE_TDM
  if (wave == 0) __builtin_amdgcn_s_wait_tensorcnt(0);
#endif
  __syncthreads();

  const int nIter = K >> 5;           // K is a multiple of 32
  for (int it = 0; it < nIter; ++it) {
    const int cur = it & 1, nxt = cur ^ 1;
    const bool more = (it + 1) < nIter;

    if (more) {                       // prefetch tile i+1 into other buffer
      issueB((it + 1) << 5, nxt);
      stageA((it + 1) << 5, nxt);
    }

    // --- A fragment: two contiguous b128 runs per lane, reused x4 ---
    union { v16bf v; uint4 q[2]; } a;
    {
      const unsigned short* pr = &lA[cur][wm * 16 + l16][0];
      a.q[0] = *(const uint4*)(pr + 8 * half);        // k = 8h .. 8h+7
      a.q[1] = *(const uint4*)(pr + 16 + 8 * half);   // k = 16+8h ..
    }
    // --- B fragments: two b128 runs per lane (k = 8h.., 16+8h..) ---
    union { v16bf v; uint4 q[2]; } b[4];
#pragma unroll
    for (int j = 0; j < 4; ++j) {
      const unsigned short* p = &lB[cur][wn * 64 + j * 16 + l16][0];
      b[j].q[0] = *(const uint4*)(p + 8 * half);
      b[j].q[1] = *(const uint4*)(p + 16 + 8 * half);
    }
#pragma unroll
    for (int j = 0; j < 4; ++j)
      c[j] = __builtin_amdgcn_wmma_f32_16x16x32_bf16(false, a.v, false, b[j].v,
                                                     (short)0, c[j], false, false);

#if USE_TDM
    if (more && wave == 0) __builtin_amdgcn_s_wait_tensorcnt(0);
#endif
    __syncthreads();                  // publish nxt, retire cur
  }

  float bv[4] = {0.f, 0.f, 0.f, 0.f};
  if (bias) {
#pragma unroll
    for (int j = 0; j < 4; ++j) bv[j] = bias[nBase + j * 16 + l16];
  }
#pragma unroll
  for (int r = 0; r < 8; ++r) {
    int m = mBase + r + 8 * half;
    if (m < M) {
#pragma unroll
      for (int j = 0; j < 4; ++j) {
        float v = c[j][r] + bv[j];
        if (flags & 2) v = fmaxf(v, 0.f);
        C[(size_t)m * ldc + nBase + j * 16 + l16] = v;
      }
    }
  }
}

// ---------------------------------------------------------------------------
// Utility / scatter kernels
// ---------------------------------------------------------------------------
__global__ void k_fill(float* p, float v, long long n) {
  long long i = (long long)blockIdx.x * blockDim.x + threadIdx.x;
  long long s = (long long)gridDim.x * blockDim.x;
  for (; i < n; i += s) p[i] = v;
}

// convert f32 weight [K,N] -> bf16 TRANSPOSED [N,K]
__global__ void k_conv_T(const float* __restrict__ s,
                         unsigned short* __restrict__ d, int K, int N) {
  int i = blockIdx.x * blockDim.x + threadIdx.x;
  if (i < K * N) {
    int k = i / N, n = i % N;
    d[(size_t)n * K + k] = f2bf(s[i]);
  }
}

__global__ void k_edge_count(const int* __restrict__ dst, float* cnt, int E) {
  int e = blockIdx.x * blockDim.x + threadIdx.x;
  if (e < E) atomicAdd(&cnt[dst[e]], 1.0f);
}

__global__ void k_edge_count_rel(const int* __restrict__ dst,
                                 const int* __restrict__ et, float* cnt, int E) {
  int e = blockIdx.x * blockDim.x + threadIdx.x;
  if (e < E) atomicAdd(&cnt[dst[e] * GR + et[e]], 1.0f);
}

// one block per edge, thread per feature dim
__global__ void k_scatter_add(const float* __restrict__ x,
                              const int* __restrict__ src,
                              const int* __restrict__ dst,
                              float* __restrict__ acc, int D, int E) {
  int e = blockIdx.x;
  if (e >= E) return;
  int s = src[e], t = dst[e], d = threadIdx.x;
  if (d < D)
    atomicAdd(&acc[(size_t)t * D + d], x[(size_t)s * D + d]);
}

__global__ void k_scatter_add_rel(const float* __restrict__ x,
                                  const int* __restrict__ src,
                                  const int* __restrict__ dst,
                                  const int* __restrict__ et,
                                  float* __restrict__ acc, int E) {
  int e = blockIdx.x;
  if (e >= E) return;
  int s = src[e], seg = dst[e] * GR + et[e], d = threadIdx.x;
  atomicAdd(&acc[(size_t)seg * GH + d], x[(size_t)s * GH + d]);
}

__global__ void k_div_cnt(float* __restrict__ acc,
                          const float* __restrict__ cnt, int D) {
  int row = blockIdx.x, d = threadIdx.x;
  if (d < D)
    acc[(size_t)row * D + d] /= fmaxf(cnt[row], 1.0f);
}

// ---------------------------------------------------------------------------
// GAT kernels
// ---------------------------------------------------------------------------
__global__ void k_gat_al(const float* __restrict__ h,
                         const float* __restrict__ attS,
                         const float* __restrict__ attD,
                         float* __restrict__ als, float* __restrict__ ald,
                         int n) {
  int i = blockIdx.x * blockDim.x + threadIdx.x;
  if (i >= n * GHEADS) return;
  int node = i >> 1, hh = i & 1;
  const float* hp = h + (size_t)node * (GHEADS * GH) + hh * GH;
  float ss = 0.f, sd = 0.f;
  for (int d = 0; d < GH; ++d) {
    float hv = hp[d];
    ss += hv * attS[hh * GH + d];
    sd += hv * attD[hh * GH + d];
  }
  als[node * GHEADS + hh] = ss;
  ald[node * GHEADS + hh] = sd;
}

__global__ void k_gat_max(const int* __restrict__ src,
                          const int* __restrict__ dst,
                          const float* __restrict__ als,
                          const float* __restrict__ ald,
                          float* __restrict__ m, int E, int n) {
  int e = blockIdx.x * blockDim.x + threadIdx.x;
  if (e >= E + n) return;
  int s, t;
  if (e < E) { s = src[e]; t = dst[e]; } else { s = t = e - E; }
#pragma unroll
  for (int hh = 0; hh < GHEADS; ++hh) {
    float v = als[s * GHEADS + hh] + ald[t * GHEADS + hh];
    v = (v > 0.f) ? v : 0.2f * v;        // leaky_relu(0.2)
    atomicMaxF(&m[t * GHEADS + hh], v);
  }
}

// one block per edge (incl self-loops), thread per dim
__global__ void k_gat_accum(const int* __restrict__ src,
                            const int* __restrict__ dst,
                            const float* __restrict__ als,
                            const float* __restrict__ ald,
                            const float* __restrict__ m,
                            const float* __restrict__ h,
                            float* __restrict__ den,
                            float* __restrict__ acc, int E, int n) {
  int e = blockIdx.x;
  if (e >= E + n) return;
  int s, t;
  if (e < E) { s = src[e]; t = dst[e]; } else { s = t = e - E; }
  int d = threadIdx.x;
#pragma unroll
  for (int hh = 0; hh < GHEADS; ++hh) {
    float v = als[s * GHEADS + hh] + ald[t * GHEADS + hh];
    v = (v > 0.f) ? v : 0.2f * v;
    float ex = __expf(v - m[t * GHEADS + hh]);
    if (d == 0) atomicAdd(&den[t * GHEADS + hh], ex);
    atomicAdd(&acc[(size_t)t * (GHEADS * GH) + hh * GH + d],
              ex * h[(size_t)s * (GHEADS * GH) + hh * GH + d]);
  }
}

__global__ void k_gat_final(const float* __restrict__ acc,
                            const float* __restrict__ den,
                            const float* __restrict__ b,
                            float* __restrict__ xg) {
  int node = blockIdx.x, d = threadIdx.x;
  float s = 0.f;
#pragma unroll
  for (int hh = 0; hh < GHEADS; ++hh)
    s += acc[(size_t)node * (GHEADS * GH) + hh * GH + d] /
         fmaxf(den[node * GHEADS + hh], 1e-16f);
  xg[(size_t)node * GH + d] = fmaxf(s * (1.0f / GHEADS) + b[d], 0.f);
}

// ---------------------------------------------------------------------------
// L2-normalize rows of 128: one wave (32 lanes) per row, 4 elems/lane
// ---------------------------------------------------------------------------
__global__ void k_norm(const float* __restrict__ in, float* __restrict__ out,
                       int n) {
  int wave = threadIdx.x >> 5, lane = threadIdx.x & 31;
  int row = blockIdx.x * 8 + wave;
  if (row >= n) return;
  float v[4]; float s = 0.f;
#pragma unroll
  for (int i = 0; i < 4; ++i) {
    v[i] = in[(size_t)row * GEMB + lane + i * 32];
    s += v[i] * v[i];
  }
#pragma unroll
  for (int off = 16; off > 0; off >>= 1) s += __shfl_xor(s, off, 32);
  float inv = 1.0f / fmaxf(sqrtf(s), 1e-12f);
#pragma unroll
  for (int i = 0; i < 4; ++i)
    out[(size_t)row * GEMB + lane + i * 32] = v[i] * inv;
}

// ---------------------------------------------------------------------------
extern "C" void kernel_launch(void* const* d_in, const int* in_sizes, int n_in,
                              void* d_out, int out_size, void* d_ws, size_t ws_size,
                              hipStream_t stream) {
  (void)in_sizes; (void)n_in; (void)out_size; (void)ws_size;
  const float* x     = (const float*)d_in[0];
  const int*   ei    = (const int*)d_in[1];
  const int*   et    = (const int*)d_in[2];
  const float* s1Wl  = (const float*)d_in[3];
  const float* s1Wr  = (const float*)d_in[4];
  const float* s1b   = (const float*)d_in[5];
  const float* s2Wl  = (const float*)d_in[6];
  const float* s2Wr  = (const float*)d_in[7];
  const float* s2b   = (const float*)d_in[8];
  const float* gatW  = (const float*)d_in[9];
  const float* attS  = (const float*)d_in[10];
  const float* attD  = (const float*)d_in[11];
  const float* gatb  = (const float*)d_in[12];
  const float* Wrel  = (const float*)d_in[13];
  const float* Wroot = (const float*)d_in[14];
  const float* rgb   = (const float*)d_in[15];
  const float* fW    = (const float*)d_in[16];
  const float* fb    = (const float*)d_in[17];

  const int* src = ei;
  const int* dst = ei + GE;

  // ---- workspace layout (floats, 16B aligned) ----
  float* ws = (float*)d_ws;
  size_t o = 0;
  auto take = [&](size_t nElem) { size_t r = o; o += (nElem + 3) & ~(size_t)3; return r; };
  float* cnt  = ws + take((size_t)GN * GR);
  float* agg1 = ws + take((size_t)GN * GIN);
  float* x1   = ws + take((size_t)GN * GH);
  float* agg2 = ws + take((size_t)GN * GH);
  float* xs   = ws + take((size_t)GN * GH);            // x_sage
  float* hmat = ws + take((size_t)GN * GHEADS * GH);   // GAT projection
  float* als  = ws + take((size_t)GN * GHEADS);
  float* ald  = ws + take((size_t)GN * GHEADS);
  float* mmax = ws + take((size_t)GN * GHEADS);
  float* den  = ws + take((size_t)GN * GHEADS);
  float* gacc = ws + take((size_t)GN * GHEADS * GH);
  float* xg   = ws + take((size_t)GN * GH);            // x_gat
  float* ragg = ws + take((size_t)GN * GR * GH);
  float* rout = ws + take((size_t)GN * GEMB);          // x_rgcn
  float* fus  = ws + take((size_t)GN * GEMB);

  // bf16 TRANSPOSED weights ([N][K] layout)
  unsigned short* wb = (unsigned short*)(ws + o);
  size_t wo = 0;
  auto wtake = [&](size_t nElem) { size_t r = wo; wo += (nElem + 7) & ~(size_t)7; return r; };
  unsigned short* bWl1  = wb + wtake((size_t)GIN * GH);
  unsigned short* bWr1  = wb + wtake((size_t)GIN * GH);
  unsigned short* bWl2  = wb + wtake((size_t)GH * GH);
  unsigned short* bWr2  = wb + wtake((size_t)GH * GH);
  unsigned short* bGat  = wb + wtake((size_t)GH * GHEADS * GH);
  unsigned short* bRel  = wb + wtake((size_t)GR * GH * GEMB);
  unsigned short* bRoot = wb + wtake((size_t)GH * GEMB);
  unsigned short* bFus  = wb + wtake((size_t)(GH + GH + GEMB) * GEMB);

  auto fill = [&](float* p, float v, size_t nElem) {
    k_fill<<<1024, 256, 0, stream>>>(p, v, (long long)nElem);
  };
  auto convT = [&](const float* s_, unsigned short* d_, int K_, int N_) {
    int n = K_ * N_;
    k_conv_T<<<(n + 255) / 256, 256, 0, stream>>>(s_, d_, K_, N_);
  };
  auto gemm = [&](const float* A, int lda, const unsigned short* Bt, int ldb,
                  const float* bias, float* C, int ldc,
                  int M, int K, int N, int flags) {
    dim3 g(N / 128, (M + 63) / 64);   // N is a multiple of 128 in all calls
    k_wmma_gemm<<<g, 256, 0, stream>>>(A, lda, Bt, ldb, bias, C, ldc, M, K, N, flags);
  };

  // ---- weights -> bf16 transposed ----
  convT(s1Wl, bWl1, GIN, GH);
  convT(s1Wr, bWr1, GIN, GH);
  convT(s2Wl, bWl2, GH, GH);
  convT(s2Wr, bWr2, GH, GH);
  convT(gatW, bGat, GH, GHEADS * GH);
  for (int r = 0; r < GR; ++r)
    convT(Wrel + (size_t)r * GH * GEMB, bRel + (size_t)r * GEMB * GH, GH, GEMB);
  convT(Wroot, bRoot, GH, GEMB);
  convT(fW, bFus, GH + GH + GEMB, GEMB);   // -> [128][640]

  // ---- SAGE layer 1: x1 = relu(segmean(x)@Wl + x@Wr + b) ----
  fill(cnt, 0.f, GN);
  fill(agg1, 0.f, (size_t)GN * GIN);
  k_edge_count<<<(GE + 255) / 256, 256, 0, stream>>>(dst, cnt, GE);
  k_scatter_add<<<GE, GIN, 0, stream>>>(x, src, dst, agg1, GIN, GE);
  k_div_cnt<<<GN, GIN, 0, stream>>>(agg1, cnt, GIN);
  gemm(agg1, GIN, bWl1, GIN, s1b, x1, GH, GN, GIN, GH, 0);
  gemm(x,    GIN, bWr1, GIN, nullptr, x1, GH, GN, GIN, GH, 3);  // accum + relu

  // ---- SAGE layer 2 ----
  fill(cnt, 0.f, GN);
  fill(agg2, 0.f, (size_t)GN * GH);
  k_edge_count<<<(GE + 255) / 256, 256, 0, stream>>>(dst, cnt, GE);
  k_scatter_add<<<GE, GH, 0, stream>>>(x1, src, dst, agg2, GH, GE);
  k_div_cnt<<<GN, GH, 0, stream>>>(agg2, cnt, GH);
  gemm(agg2, GH, bWl2, GH, s2b, xs, GH, GN, GH, GH, 0);
  gemm(x1,   GH, bWr2, GH, nullptr, xs, GH, GN, GH, GH, 3);

  // ---- GAT (heads=2, concat=False, self-loops) ----
  gemm(xs, GH, bGat, GH, nullptr, hmat, GHEADS * GH, GN, GH, GHEADS * GH, 0);
  k_gat_al<<<(GN * GHEADS + 255) / 256, 256, 0, stream>>>(hmat, attS, attD, als, ald, GN);
  fill(mmax, -__builtin_inff(), (size_t)GN * GHEADS);
  fill(den, 0.f, (size_t)GN * GHEADS);
  fill(gacc, 0.f, (size_t)GN * GHEADS * GH);
  k_gat_max<<<(GE + GN + 255) / 256, 256, 0, stream>>>(src, dst, als, ald, mmax, GE, GN);
  k_gat_accum<<<GE + GN, GH, 0, stream>>>(src, dst, als, ald, mmax, hmat, den, gacc, GE, GN);
  k_gat_final<<<GN, GH, 0, stream>>>(gacc, den, gatb, xg);

  // ---- RGCN: rout = sum_r segmean_r(xg) @ Wrel_r + xg @ Wroot + b ----
  fill(cnt, 0.f, (size_t)GN * GR);
  fill(ragg, 0.f, (size_t)GN * GR * GH);
  k_edge_count_rel<<<(GE + 255) / 256, 256, 0, stream>>>(dst, et, cnt, GE);
  k_scatter_add_rel<<<GE, GH, 0, stream>>>(xg, src, dst, et, ragg, GE);
  k_div_cnt<<<GN * GR, GH, 0, stream>>>(ragg, cnt, GH);
  gemm(xg, GH, bRoot, GH, rgb, rout, GEMB, GN, GH, GEMB, 0);
  for (int r = 0; r < GR; ++r)
    gemm(ragg + (size_t)r * GH, GR * GH, bRel + (size_t)r * GEMB * GH, GH,
         nullptr, rout, GEMB, GN, GH, GEMB, 1);

  // ---- fusion: out = [xs|xg|rout] @ fW + fb, then row L2-normalize ----
  // bFus is [128][640]; slice columns 0..255 / 256..511 / 512..639
  gemm(xs,   GH,   bFus,          GH + GH + GEMB, fb,      fus, GEMB, GN, GH,   GEMB, 0);
  gemm(xg,   GH,   bFus + GH,     GH + GH + GEMB, nullptr, fus, GEMB, GN, GH,   GEMB, 1);
  gemm(rout, GEMB, bFus + 2 * GH, GH + GH + GEMB, nullptr, fus, GEMB, GN, GEMB, GEMB, 1);

  k_norm<<<(GN + 7) / 8, 256, 0, stream>>>(fus, (float*)d_out, GN);
}
